// GCN_55628416418160
// MI455X (gfx1250) — matrix operand order repaired and checked by reading, not compiled, not run on hardware
//
#include <hip/hip_runtime.h>
#include <hip/hip_bf16.h>
#include <math.h>

typedef __attribute__((ext_vector_type(16))) __bf16 v16bf;
typedef __attribute__((ext_vector_type(8)))  __bf16 v8bf;
typedef __attribute__((ext_vector_type(8)))  float  v8f;

#define NN   100000
#define NE   1600000
#define FDIM 128
#define CDIM 40
#define CPAD 48
#define BN_EPS 1e-5f

// ---------------------------------------------------------------- utilities
__global__ void k_zero_i32(int* p, int n) {
  int i = blockIdx.x * blockDim.x + threadIdx.x;
  if (i < n) p[i] = 0;
}
__global__ void k_zero_f32(float* p, int n) {
  int i = blockIdx.x * blockDim.x + threadIdx.x;
  if (i < n) p[i] = 0.0f;
}

// ---------------------------------------------------------------- degree
__global__ void k_count_deg(const int* __restrict__ ei, int* __restrict__ degi, int E) {
  int e = blockIdx.x * blockDim.x + threadIdx.x;
  if (e < E) atomicAdd(&degi[ei[E + e]], 1);   // dst endpoint
}
__global__ void k_finalize_deg(const int* __restrict__ degi, float* __restrict__ dinv,
                               float* __restrict__ rdeg, int n) {
  int i = blockIdx.x * blockDim.x + threadIdx.x;
  if (i < n) {
    float d = (float)(degi[i] + 1);            // +1 self loop
    dinv[i] = rsqrtf(d);
    rdeg[i] = 1.0f / d;
  }
}

// ---------------------------------------------------------------- conversions
__global__ void k_cvt_f32_bf16(const float* __restrict__ src, __bf16* __restrict__ dst,
                               long long n) {
  long long i = (long long)blockIdx.x * blockDim.x + threadIdx.x;
  if (i < n) dst[i] = (__bf16)src[i];
}

// Pack W (Kdim x ncols, row-major) into WMMA-B fragment order:
// Wp[((ct*4 + kc)*32 + lane)*16 + e] = W[(kc*32 + (e>>3)*16 + (lane>>4)*8 + (e&7)) * ncols + ct*16 + (lane&15)]
__global__ void k_pack_weights(const float* __restrict__ W, __bf16* __restrict__ Wp,
                               int ncols, int nColTiles) {
  int idx = blockIdx.x * blockDim.x + threadIdx.x;
  int total = nColTiles * 4 * 32 * 16;
  if (idx >= total) return;
  int e    = idx & 15;
  int lane = (idx >> 4) & 31;
  int kc   = (idx >> 9) & 3;
  int ct   = idx >> 11;
  int col  = ct * 16 + (lane & 15);
  int k    = kc * 32 + (e >> 3) * 16 + (lane >> 4) * 8 + (e & 7);
  float v  = (col < ncols) ? W[k * ncols + col] : 0.0f;
  Wp[idx] = (__bf16)v;
}

// ---------------------------------------------------------------- WMMA GEMM
// C[M x ldc] = A[M x 128] (bf16, row-major) * Bpacked (f32 accumulate).
// One wave32 per 16x16 output tile. All fragment loads are hoisted ahead of
// the 4 WMMAs so the backend can clause the loads and use partial loadcnt
// waits instead of a full s_wait_loadcnt before every matrix op.
__global__ void k_gemm_bf16_wmma(const __bf16* __restrict__ A,
                                 const __bf16* __restrict__ Bp,
                                 float* __restrict__ C,
                                 int M, int ldc, int nColTiles) {
  const int lane   = threadIdx.x & 31;
  const int wave   = blockIdx.x * (blockDim.x >> 5) + (threadIdx.x >> 5);
  const int nTiles = (M >> 4) * nColTiles;
  if (wave >= nTiles) return;                      // wave-uniform guard (EXEC stays all-1s)
  const int rowTile = wave / nColTiles;
  const int colTile = wave - rowTile * nColTiles;
  const int halfSel = lane >> 4;                   // 0 or 1
  const int lane15  = lane & 15;

  const __bf16* Arow  = A + (size_t)(rowTile * 16 + lane15) * FDIM;
  const int     aOff  = halfSel * 8;               // lanes0-15: K0-7/16-23; lanes16-31: K8-15/24-31
  const __bf16* Btile = Bp + ((size_t)colTile * 4 * 32 + lane) * 16;

  v16bf af[4], bf[4];
#pragma unroll
  for (int kc = 0; kc < 4; ++kc) {
    const int k0 = kc << 5;
    v8bf alo = *(const v8bf*)(Arow + k0 + aOff);
    v8bf ahi = *(const v8bf*)(Arow + k0 + 16 + aOff);
    af[kc] = __builtin_shufflevector(alo, ahi, 0,1,2,3,4,5,6,7,8,9,10,11,12,13,14,15);
    bf[kc] = *(const v16bf*)(Btile + (size_t)kc * 32 * 16);
  }

  v8f acc = {};
#pragma unroll
  for (int kc = 0; kc < 4; ++kc)
    acc = __builtin_amdgcn_wmma_f32_16x16x32_bf16(false, af[kc], false, bf[kc],
                                                  (short)0, acc, false, false);

  // C/D layout: VGPR r -> row (halfSel*8 + r), col = lane15
  float* Cp = C + (size_t)(rowTile * 16 + halfSel * 8) * ldc + colTile * 16 + lane15;
#pragma unroll
  for (int r = 0; r < 8; ++r) Cp[(size_t)r * ldc] = acc[r];
}

// ---------------------------------------------------------------- aggregation
// a[i,f] = (1/deg_i) * h[i,f] + bias[f]   (pad columns -> 0)
__global__ void k_agg_init(const float* __restrict__ h, const float* __restrict__ rdeg,
                           const float* __restrict__ bias, float* __restrict__ a,
                           int n, int F, int ld) {
  long long idx = (long long)blockIdx.x * blockDim.x + threadIdx.x;
  long long total = (long long)n * ld;
  if (idx >= total) return;
  int f = (int)(idx % ld);
  a[idx] = (f < F) ? rdeg[idx / ld] * h[idx] + bias[f] : 0.0f;
}

// per-edge scatter, float4 per thread: a[dst,f..f+3] += dinv[s]*dinv[d]*h[src,f..f+3]
// lanesPerEdgeShift: log2(threads covering one row chunk); each thread owns 4 floats.
__global__ void k_agg_edges4(const int* __restrict__ ei, const float* __restrict__ dinv,
                             const float* __restrict__ h, float* __restrict__ a,
                             int E, int F4 /*F/4*/, int ld, int lpeShift) {
  const int lpe = 1 << lpeShift;                   // threads per edge
  const int t = threadIdx.x & (lpe - 1);           // float4 index within row
  long long e = (long long)blockIdx.x * (blockDim.x >> lpeShift) + (threadIdx.x >> lpeShift);
  if (e >= E || t >= F4) return;
  int s = ei[e];
  int d = ei[(long long)E + e];
  float nrm = dinv[s] * dinv[d];
  const float4 hv = *(const float4*)(h + (size_t)s * ld + t * 4);
  float* ap = a + (size_t)d * ld + t * 4;
  unsafeAtomicAdd(ap + 0, nrm * hv.x);
  unsafeAtomicAdd(ap + 1, nrm * hv.y);
  unsafeAtomicAdd(ap + 2, nrm * hv.z);
  unsafeAtomicAdd(ap + 3, nrm * hv.w);
}

// ---------------------------------------------------------------- batchnorm
// per-feature partial sums over 256-row chunks, atomically merged
__global__ void k_bn_stats(const float* __restrict__ a, float* __restrict__ sums,
                           float* __restrict__ sumsq, int n) {
  const int f = threadIdx.x;                       // blockDim.x == 128
  long long r0 = (long long)blockIdx.x * 256;
  float s = 0.0f, q = 0.0f;
  for (int i = 0; i < 256; ++i) {
    long long r = r0 + i;
    if (r < n) {
      float v = a[r * FDIM + f];
      s += v; q += v * v;
    }
  }
  unsafeAtomicAdd(&sums[f], s);
  unsafeAtomicAdd(&sumsq[f], q);
}

// fused BN (training-mode, biased var) + ReLU + bf16 convert for next GEMM
__global__ void k_bn_apply(const float* __restrict__ a, const float* __restrict__ sums,
                           const float* __restrict__ sumsq, const float* __restrict__ g,
                           const float* __restrict__ be, __bf16* __restrict__ xb, int n) {
  long long idx = (long long)blockIdx.x * blockDim.x + threadIdx.x;
  if (idx >= (long long)n * FDIM) return;
  int f = (int)(idx & (FDIM - 1));
  const float invN = 1.0f / (float)n;
  float mean = sums[f] * invN;
  float var  = sumsq[f] * invN - mean * mean;
  float sc   = g[f] * rsqrtf(var + BN_EPS);
  float sh   = be[f] - mean * sc;
  float y = fmaxf(a[idx] * sc + sh, 0.0f);
  xb[idx] = (__bf16)y;
}

// ---------------------------------------------------------------- log-softmax
__global__ void k_log_softmax(const float* __restrict__ a, float* __restrict__ out, int n) {
  long long row = (long long)blockIdx.x * blockDim.x + threadIdx.x;
  if (row >= n) return;
  const float4* p4 = (const float4*)(a + row * CPAD);   // 16B-aligned (CPAD=48)
  float4 v[10];
#pragma unroll
  for (int i = 0; i < 10; ++i) v[i] = p4[i];            // 40 classes = 10 float4
  float m = v[0].x;
#pragma unroll
  for (int i = 0; i < 10; ++i) {
    m = fmaxf(m, v[i].x); m = fmaxf(m, v[i].y);
    m = fmaxf(m, v[i].z); m = fmaxf(m, v[i].w);
  }
  float s = 0.0f;
#pragma unroll
  for (int i = 0; i < 10; ++i) {
    s += expf(v[i].x - m); s += expf(v[i].y - m);
    s += expf(v[i].z - m); s += expf(v[i].w - m);
  }
  float ls = m + logf(s);
  float* o = out + row * CDIM;
#pragma unroll
  for (int i = 0; i < 10; ++i) {
    o[i * 4 + 0] = v[i].x - ls; o[i * 4 + 1] = v[i].y - ls;
    o[i * 4 + 2] = v[i].z - ls; o[i * 4 + 3] = v[i].w - ls;
  }
}

// ---------------------------------------------------------------- launcher
static inline int cdiv(long long a, long long b) { return (int)((a + b - 1) / b); }

extern "C" void kernel_launch(void* const* d_in, const int* in_sizes, int n_in,
                              void* d_out, int out_size, void* d_ws, size_t ws_size,
                              hipStream_t stream) {
  const float* x   = (const float*)d_in[0];
  const float* W1  = (const float*)d_in[1];
  const float* b1  = (const float*)d_in[2];
  const float* W2  = (const float*)d_in[3];
  const float* b2  = (const float*)d_in[4];
  const float* W3  = (const float*)d_in[5];
  const float* b3  = (const float*)d_in[6];
  const float* g1  = (const float*)d_in[7];
  const float* be1 = (const float*)d_in[8];
  const float* g2  = (const float*)d_in[9];
  const float* be2 = (const float*)d_in[10];
  const int*   ei  = (const int*)d_in[11];
  float* out = (float*)d_out;

  // ---- workspace carve-out (256B aligned)
  size_t off = 0;
  char* base = (char*)d_ws;
  auto carve = [&](size_t bytes) -> void* {
    void* p = base + off;
    off += (bytes + 255) & ~(size_t)255;
    return p;
  };
  int*    degi  = (int*)   carve((size_t)NN * 4);
  float*  dinv  = (float*) carve((size_t)NN * 4);
  float*  rdeg  = (float*) carve((size_t)NN * 4);
  float*  stats = (float*) carve(256 * 4);                 // sums[128] | sumsq[128]
  __bf16* Wp1   = (__bf16*)carve((size_t)8 * 4 * 32 * 16 * 2);
  __bf16* Wp2   = (__bf16*)carve((size_t)8 * 4 * 32 * 16 * 2);
  __bf16* Wp3   = (__bf16*)carve((size_t)3 * 4 * 32 * 16 * 2);
  __bf16* xb    = (__bf16*)carve((size_t)NN * FDIM * 2);
  float*  bufH  = (float*) carve((size_t)NN * FDIM * 4);
  float*  bufA  = (float*) carve((size_t)NN * FDIM * 4);
  float* sums  = stats;
  float* sumsq = stats + 128;

  const long long nf = (long long)NN * FDIM;

  // ---- degree / normalization
  k_zero_i32<<<cdiv(NN, 256), 256, 0, stream>>>(degi, NN);
  k_count_deg<<<cdiv(NE, 256), 256, 0, stream>>>(ei, degi, NE);
  k_finalize_deg<<<cdiv(NN, 256), 256, 0, stream>>>(degi, dinv, rdeg, NN);

  // ---- input + weight conversion/packing
  k_cvt_f32_bf16<<<cdiv(nf, 256), 256, 0, stream>>>(x, xb, nf);
  k_pack_weights<<<cdiv(8 * 2048, 256), 256, 0, stream>>>(W1, Wp1, FDIM, 8);
  k_pack_weights<<<cdiv(8 * 2048, 256), 256, 0, stream>>>(W2, Wp2, FDIM, 8);
  k_pack_weights<<<cdiv(3 * 2048, 256), 256, 0, stream>>>(W3, Wp3, CDIM, 3);

  const int gemmBlocksH = cdiv((NN / 16) * 8, 8);   // 8 waves/block of 256 thr
  const int gemmBlocksC = cdiv((NN / 16) * 3, 8);

  // edge-scatter grids: float4 per thread
  //   layers 1-2: 32 lanes/edge -> 8 edges per 256-block
  //   layer 3   : 16 lanes/edge (10 active) -> 16 edges per 256-block
  const int edgeBlocksH = cdiv((long long)NE * 32, 256);
  const int edgeBlocksC = cdiv((long long)NE * 16, 256);

  // ================= layer 1 =================
  k_gemm_bf16_wmma<<<gemmBlocksH, 256, 0, stream>>>(xb, Wp1, bufH, NN, FDIM, 8);
  k_agg_init<<<cdiv(nf, 256), 256, 0, stream>>>(bufH, rdeg, b1, bufA, NN, FDIM, FDIM);
  k_agg_edges4<<<edgeBlocksH, 256, 0, stream>>>(ei, dinv, bufH, bufA, NE, 32, FDIM, 5);
  k_zero_f32<<<1, 256, 0, stream>>>(stats, 256);
  k_bn_stats<<<cdiv(NN, 256), 128, 0, stream>>>(bufA, sums, sumsq, NN);
  k_bn_apply<<<cdiv(nf, 256), 256, 0, stream>>>(bufA, sums, sumsq, g1, be1, xb, NN);

  // ================= layer 2 =================
  k_gemm_bf16_wmma<<<gemmBlocksH, 256, 0, stream>>>(xb, Wp2, bufH, NN, FDIM, 8);
  k_agg_init<<<cdiv(nf, 256), 256, 0, stream>>>(bufH, rdeg, b2, bufA, NN, FDIM, FDIM);
  k_agg_edges4<<<edgeBlocksH, 256, 0, stream>>>(ei, dinv, bufH, bufA, NE, 32, FDIM, 5);
  k_zero_f32<<<1, 256, 0, stream>>>(stats, 256);
  k_bn_stats<<<cdiv(NN, 256), 128, 0, stream>>>(bufA, sums, sumsq, NN);
  k_bn_apply<<<cdiv(nf, 256), 256, 0, stream>>>(bufA, sums, sumsq, g2, be2, xb, NN);

  // ================= layer 3 =================
  k_gemm_bf16_wmma<<<gemmBlocksC, 256, 0, stream>>>(xb, Wp3, bufH, NN, CPAD, 3);
  k_agg_init<<<cdiv((long long)NN * CPAD, 256), 256, 0, stream>>>(bufH, rdeg, b3, bufA,
                                                                  NN, CDIM, CPAD);
  k_agg_edges4<<<edgeBlocksC, 256, 0, stream>>>(ei, dinv, bufH, bufA, NE, 10, CPAD, 4);
  k_log_softmax<<<cdiv(NN, 256), 256, 0, stream>>>(bufA, out, NN);
  (void)in_sizes; (void)n_in; (void)out_size; (void)ws_size;
}
